// FlowmatchingActionHead_14078902796564
// MI455X (gfx1250) — matrix operand order repaired and checked by reading, not compiled, not run on hardware
//
#include <hip/hip_runtime.h>

// ---------------------------------------------------------------------------
// Fused per-category 2-layer MLP for MI455X (gfx1250, wave32, WMMA bf16).
//   hidden = relu(x @ W1[cat] + b1[cat]);  out = hidden @ W2[cat] + b2[cat]
// x:[64,1024,1536] f32, W1:[32,1536,1024], W2:[32,1024,1024], out:[64,1024,1024]
// bf16 inputs + f32 WMMA accumulation; hidden lives entirely in LDS.
// 32x64 wave tiles (8 WMMAs/step/wave, B fragments read exactly once),
// double-buffered weight chunks staged in two halves (b64-packed DS stores),
// software-pipelined against the WMMA chain.
// ---------------------------------------------------------------------------

typedef __bf16 bf16_t;
typedef __attribute__((ext_vector_type(16))) __bf16 v16bf;
typedef __attribute__((ext_vector_type(8)))  __bf16 v8bf;
typedef __attribute__((ext_vector_type(4)))  __bf16 v4bf;
typedef __attribute__((ext_vector_type(8)))  float  v8f;

#define DEV __device__ __forceinline__

constexpr int IN_DIM  = 1536;
constexpr int HID_DIM = 1024;
constexpr int OUT_DIM = 1024;
constexpr int B_      = 64;
constexpr int T_      = 1024;

constexpr int TTILE   = 32;            // rows of T per workgroup
constexpr int NCHUNK  = 512;           // output columns per chunk (8 waves x 64)
constexpr int KSTEP   = 32;            // K per WMMA

// LDS layout (bf16 elements), padded strides to avoid bank conflicts
constexpr int XS_STRIDE = IN_DIM + 8;   // 1544 (16B-aligned row stride)
constexpr int HS_STRIDE = HID_DIM + 8;  // 1032
constexpr int WT_STRIDE = KSTEP + 8;    // 40
constexpr int WT_ELEMS  = NCHUNK * WT_STRIDE; // one weight buffer (20480 elems)

constexpr size_t XS_BYTES = (size_t)TTILE * XS_STRIDE * sizeof(bf16_t); //  98816
constexpr size_t HS_BYTES = (size_t)TTILE * HS_STRIDE * sizeof(bf16_t); //  66048
constexpr size_t WT_BYTES = 2 * (size_t)WT_ELEMS * sizeof(bf16_t);      //  81920
constexpr size_t SMEM_BYTES = XS_BYTES + HS_BYTES + WT_BYTES;           // 246784 < 320KB

struct alignas(16) F4 { float v[4]; };

// Assemble a 16-element bf16 fragment from two aligned 16B LDS runs.
DEV v16bf load_frag16(const bf16_t* p0, const bf16_t* p1) {
  v8bf lo = *(const v8bf*)p0;
  v8bf hi = *(const v8bf*)p1;
  return __builtin_shufflevector(lo, hi, 0,1,2,3,4,5,6,7,8,9,10,11,12,13,14,15);
}

// ---- weight-chunk staging: one [KSTEP x 256] half-chunk per call ----
// Thread slices: s = tid + 256*i (i=0..1): ka = (s&7)*4 (4 k-rows), nn = (s>>3)*4.
// Each slice = 4 global float4 loads (contiguous n) packed into 4 ds_store_b64
// (contiguous k in the transposed LDS buffer).
DEV void load_w_half(F4 r[8], const float* __restrict__ Wg,
                     int k0, int nbase, int tid) {
#pragma unroll
  for (int i = 0; i < 2; ++i) {
    const int s  = tid + 256 * i;
    const int ka = (s & 7) << 2;
    const int nn = (s >> 3) << 2;
#pragma unroll
    for (int j = 0; j < 4; ++j)
      r[i * 4 + j] = *(const F4*)(Wg + (size_t)(k0 + ka + j) * 1024 + nbase + nn);
  }
}

DEV void store_w_half(bf16_t* __restrict__ buf, const F4 r[8],
                      int nb_local, int tid) {
#pragma unroll
  for (int i = 0; i < 2; ++i) {
    const int s  = tid + 256 * i;
    const int ka = (s & 7) << 2;
    const int nn = ((s >> 3) << 2) + nb_local;
#pragma unroll
    for (int cc = 0; cc < 4; ++cc) {
      v4bf p;
      p[0] = (bf16_t)r[i * 4 + 0].v[cc];
      p[1] = (bf16_t)r[i * 4 + 1].v[cc];
      p[2] = (bf16_t)r[i * 4 + 2].v[cc];
      p[3] = (bf16_t)r[i * 4 + 3].v[cc];
      *(v4bf*)(buf + (nn + cc) * WT_STRIDE + ka) = p;   // 8B, aligned, conflict-free
    }
  }
}

// acc[c*2+ar] += A[32 x K] @ W[K x 512-chunk] for this wave's 64-column strip.
DEV void gemm_chunk(const bf16_t* __restrict__ A, int a_stride, int ksteps,
                    const float* __restrict__ Wg, int n0,
                    bf16_t* __restrict__ WtBase,
                    int tid, int l16, int half, int wcol, v8f acc[8]) {
  F4 r[8];
  // Prologue: stage chunk 0 (both halves) into buffer 0.
  load_w_half(r, Wg, 0, n0, tid);
  store_w_half(WtBase, r, 0, tid);
  load_w_half(r, Wg, 0, n0 + 256, tid);
  store_w_half(WtBase, r, 256, tid);
  __syncthreads();

  for (int ks = 0; ks < ksteps; ++ks) {
    const int k0 = ks * KSTEP;
    const bf16_t* cur = WtBase + (ks & 1) * WT_ELEMS;
    bf16_t*       nxt = WtBase + ((ks + 1) & 1) * WT_ELEMS;
    const bool more = (ks + 1 < ksteps);

    // Next chunk, half 0: global loads in flight behind WMMA group 1.
    if (more) load_w_half(r, Wg, k0 + KSTEP, n0, tid);

    // Distance-2 prefetch into L2 (clamped, uniform control flow).
    {
      int kpf = k0 + 2 * KSTEP;
      if (kpf > (ksteps - 1) * KSTEP) kpf = (ksteps - 1) * KSTEP;
      __builtin_prefetch(
          Wg + (size_t)(kpf + ((tid & 7) << 2)) * 1024 + n0 + ((tid >> 3) << 2),
          0, 1);
    }

    // A fragments: both 16-row blocks (rows 0..31), reused by all 8 WMMAs.
    const bf16_t* ap0 = A + (l16)      * a_stride + k0 + half * 8;
    const bf16_t* ap1 = A + (16 + l16) * a_stride + k0 + half * 8;
    const v16bf a0 = load_frag16(ap0, ap0 + 16);
    const v16bf a1 = load_frag16(ap1, ap1 + 16);

    // WMMA group 1: column tiles c=0,1.
#pragma unroll
    for (int c = 0; c < 2; ++c) {
      const bf16_t* bp = cur + (wcol + c * 16 + l16) * WT_STRIDE + half * 8;
      const v16bf bfrag = load_frag16(bp, bp + 16);
      acc[c * 2 + 0] = __builtin_amdgcn_wmma_f32_16x16x32_bf16(
          false, a0, false, bfrag, (short)0, acc[c * 2 + 0], false, false);
      acc[c * 2 + 1] = __builtin_amdgcn_wmma_f32_16x16x32_bf16(
          false, a1, false, bfrag, (short)0, acc[c * 2 + 1], false, false);
    }

    // Drain half 0 into the other buffer; issue half 1 loads behind group 2.
    if (more) {
      store_w_half(nxt, r, 0, tid);
      load_w_half(r, Wg, k0 + KSTEP, n0 + 256, tid);
    }

    // WMMA group 2: column tiles c=2,3.
#pragma unroll
    for (int c = 2; c < 4; ++c) {
      const bf16_t* bp = cur + (wcol + c * 16 + l16) * WT_STRIDE + half * 8;
      const v16bf bfrag = load_frag16(bp, bp + 16);
      acc[c * 2 + 0] = __builtin_amdgcn_wmma_f32_16x16x32_bf16(
          false, a0, false, bfrag, (short)0, acc[c * 2 + 0], false, false);
      acc[c * 2 + 1] = __builtin_amdgcn_wmma_f32_16x16x32_bf16(
          false, a1, false, bfrag, (short)0, acc[c * 2 + 1], false, false);
    }

    if (more) store_w_half(nxt, r, 256, tid);
    __syncthreads();
  }
}

__global__ __launch_bounds__(256)
void cat_mlp_fused_kernel(const float* __restrict__ x,
                          const int*   __restrict__ cat_ids,
                          const float* __restrict__ W1,
                          const float* __restrict__ b1,
                          const float* __restrict__ W2,
                          const float* __restrict__ b2,
                          float*       __restrict__ out) {
  extern __shared__ char smem[];
  bf16_t* Xs = (bf16_t*)smem;                         // [TTILE][XS_STRIDE]
  bf16_t* Hs = (bf16_t*)(smem + XS_BYTES);            // [TTILE][HS_STRIDE]
  bf16_t* Wt = (bf16_t*)(smem + XS_BYTES + HS_BYTES); // [2][NCHUNK][WT_STRIDE]

  const int tid  = threadIdx.x;
  const int lane = tid & 31;
  const int wv   = tid >> 5;       // wave id 0..7
  const int l16  = lane & 15;      // N / M index within fragment
  const int half = lane >> 4;      // K-phase selector for A/B fragments
  const int wcol = wv << 6;        // wave column strip: 0,64,...,448

  const int bb = blockIdx.y;
  const int t0 = blockIdx.x * TTILE;
  const int cat = cat_ids[bb];

  const float* W1c = W1 + (size_t)cat * IN_DIM  * HID_DIM;
  const float* W2c = W2 + (size_t)cat * HID_DIM * OUT_DIM;
  const float* b1c = b1 + (size_t)cat * HID_DIM;
  const float* b2c = b2 + (size_t)cat * OUT_DIM;
  const float* xrow0 = x   + ((size_t)bb * T_ + t0) * IN_DIM;
  float*       orow0 = out + ((size_t)bb * T_ + t0) * OUT_DIM;

  // ---- Stage x tile [32 x 1536] fp32 -> bf16 LDS (kept resident) ----
  for (int idx = tid * 4; idx < TTILE * IN_DIM; idx += 256 * 4) {
    const int row = idx / IN_DIM;
    const int k   = idx - row * IN_DIM;                // multiple of 4
    const F4 v = *(const F4*)(xrow0 + (size_t)row * IN_DIM + k);
    bf16_t* d = Xs + row * XS_STRIDE + k;
    v4bf p;
    p[0] = (bf16_t)v.v[0]; p[1] = (bf16_t)v.v[1];
    p[2] = (bf16_t)v.v[2]; p[3] = (bf16_t)v.v[3];
    *(v4bf*)d = p;
  }
  __syncthreads();

  // =================== Layer 1: Hs = relu(Xs @ W1 + b1) ===================
  for (int nc = 0; nc < HID_DIM / NCHUNK; ++nc) {
    const int n0 = nc * NCHUNK;
    v8f acc[8] = {};
    gemm_chunk(Xs, XS_STRIDE, IN_DIM / KSTEP, W1c, n0, Wt,
               tid, l16, half, wcol, acc);
#pragma unroll
    for (int c = 0; c < 4; ++c) {
      const int n  = n0 + wcol + c * 16 + l16;   // lane-invariant across VGPRs
      const float bv = b1c[n];
#pragma unroll
      for (int ar = 0; ar < 2; ++ar)
#pragma unroll
        for (int v = 0; v < 8; ++v) {
          const int m = ar * 16 + half * 8 + v;  // CDNA5 f32 C/D layout
          const float h = acc[c * 2 + ar][v] + bv;
          Hs[m * HS_STRIDE + n] = (bf16_t)(h > 0.f ? h : 0.f);
        }
    }
  }

  // =================== Layer 2: out = Hs @ W2 + b2 ===================
  for (int nc = 0; nc < OUT_DIM / NCHUNK; ++nc) {
    const int n0 = nc * NCHUNK;
    v8f acc[8] = {};
    gemm_chunk(Hs, HS_STRIDE, HID_DIM / KSTEP, W2c, n0, Wt,
               tid, l16, half, wcol, acc);
#pragma unroll
    for (int c = 0; c < 4; ++c) {
      const int n  = n0 + wcol + c * 16 + l16;
      const float bv = b2c[n];
#pragma unroll
      for (int ar = 0; ar < 2; ++ar)
#pragma unroll
        for (int v = 0; v < 8; ++v) {
          const int m = ar * 16 + half * 8 + v;
          orow0[(size_t)m * OUT_DIM + n] = acc[c * 2 + ar][v] + bv;
        }
    }
  }
}

extern "C" void kernel_launch(void* const* d_in, const int* in_sizes, int n_in,
                              void* d_out, int out_size, void* d_ws, size_t ws_size,
                              hipStream_t stream) {
  (void)in_sizes; (void)n_in; (void)out_size; (void)d_ws; (void)ws_size;
  const float* x       = (const float*)d_in[0];
  const int*   cat_ids = (const int*)  d_in[1];
  const float* W1      = (const float*)d_in[2];
  const float* b1      = (const float*)d_in[3];
  const float* W2      = (const float*)d_in[4];
  const float* b2      = (const float*)d_in[5];
  float*       out     = (float*)d_out;

  static_assert(SMEM_BYTES <= 320 * 1024, "LDS budget exceeded");
  hipFuncSetAttribute((const void*)cat_mlp_fused_kernel,
                      hipFuncAttributeMaxDynamicSharedMemorySize, (int)SMEM_BYTES);

  dim3 grid(T_ / TTILE, B_);   // 32 x 64 = 2048 workgroups
  dim3 block(256);             // 8 wave32s
  cat_mlp_fused_kernel<<<grid, block, (int)SMEM_BYTES, stream>>>(
      x, cat_ids, W1, b1, W2, b2, out);
}
